// Model_22797686407666
// MI455X (gfx1250) — compile-verified, matrix operand order checked by reading, not run on hardware
//
#include <hip/hip_runtime.h>
#include <hip/hip_bf16.h>

typedef __attribute__((ext_vector_type(16))) _Float16 v16h;
typedef __attribute__((ext_vector_type(8)))  _Float16 v8h;
typedef __attribute__((ext_vector_type(8)))  float    v8f;

// ---------------------------------------------------------------- utilities

__global__ void zero_f32(float* __restrict__ p, long n) {
  long t = (long)blockIdx.x * blockDim.x + threadIdx.x;
  if (t < n) p[t] = 0.0f;
}

__global__ void degree_count(const int* __restrict__ src, const int* __restrict__ dst,
                             int E, unsigned* __restrict__ degS, unsigned* __restrict__ degD) {
  int t = blockIdx.x * blockDim.x + threadIdx.x;
  if (t < E) {
    atomicAdd(&degS[src[t]], 1u);
    atomicAdd(&degD[dst[t]], 1u);
  }
}

__global__ void deg_to_rsqrt(const unsigned* __restrict__ deg, float* __restrict__ rs, int n) {
  int t = blockIdx.x * blockDim.x + threadIdx.x;
  if (t < n) {
    unsigned c = deg[t];
    rs[t] = rsqrtf((float)(c ? c : 1u));   // clip(deg,1) then rsqrt
  }
}

// --------------------------------------------------------- weight packing
// Convert B[K,N] f32 (row-major, row stride N) into f16 WMMA B-fragments,
// zero-padded in both K (to 32*ksteps) and N (to 16*ntiles).  Fragment for
// (tile nt, kstep ks, lane l) is stored at v16h index ((nt*ksteps+ks)*32+l).
// Layout per ISA: lane supplies column n = l&15; element e holds
// k = ks*32 + (e>>3)*16 + (l>>4)*8 + (e&7).

__global__ void pack_b_f16(const float* __restrict__ B, int K, int N,
                           _Float16* __restrict__ Bp, int ksteps) {
  const int t = blockIdx.x * blockDim.x + threadIdx.x;
  const int lane = t & 31;
  const int ks = (t >> 5) % ksteps;
  const int nt = (t >> 5) / ksteps;
  const int m16 = lane & 15, hi = lane >> 4;
  const int n = nt * 16 + m16;
  v16h v = {};
#pragma unroll
  for (int e = 0; e < 16; ++e) {
    const int k = ks * 32 + (e >> 3) * 16 + hi * 8 + (e & 7);
    if (k < K && n < N) v[e] = (_Float16)B[k * N + n];
  }
  ((v16h*)Bp)[t] = v;
}

// ------------------------------------------------- small WMMA GEMM (f32 A/C)
// C[M, 16*gridX] = (diag(rowscale) * A[M, 32*ksteps]) @ Bpacked (+ bias).
// A must be K-dense (zero-padded cols OK).  One wave per 16x16 C tile, no
// guards in the k-loop: A = 4x float4 loads, B = one v16h fragment load.

__global__ void gemm_wmma(const float* __restrict__ A, int lda,
                          const float* __restrict__ rowscale,
                          const _Float16* __restrict__ Bp, int ksteps,
                          const float* __restrict__ bias,
                          float* __restrict__ C, int ldc,
                          int M, int N) {
  const int lane = threadIdx.x & 31;
  const int m16 = lane & 15, hi = lane >> 4;
  const int tileM = blockIdx.y * 16, tileN = blockIdx.x * 16;
  const int row = tileM + m16;
  const float sA = (rowscale != nullptr) ? rowscale[row] : 1.0f;

  const float* Ap = A + (size_t)row * lda + hi * 8;
  const v16h* Bfrag = (const v16h*)Bp + ((size_t)blockIdx.x * ksteps) * 32 + lane;

  v8f acc = {};
  for (int ks = 0; ks < ksteps; ++ks) {
    const float4 a0 = *(const float4*)(Ap + ks * 32 + 0);
    const float4 a1 = *(const float4*)(Ap + ks * 32 + 4);
    const float4 a2 = *(const float4*)(Ap + ks * 32 + 16);
    const float4 a3 = *(const float4*)(Ap + ks * 32 + 20);
    v16h a;
    a[0]  = (_Float16)(a0.x * sA); a[1]  = (_Float16)(a0.y * sA);
    a[2]  = (_Float16)(a0.z * sA); a[3]  = (_Float16)(a0.w * sA);
    a[4]  = (_Float16)(a1.x * sA); a[5]  = (_Float16)(a1.y * sA);
    a[6]  = (_Float16)(a1.z * sA); a[7]  = (_Float16)(a1.w * sA);
    a[8]  = (_Float16)(a2.x * sA); a[9]  = (_Float16)(a2.y * sA);
    a[10] = (_Float16)(a2.z * sA); a[11] = (_Float16)(a2.w * sA);
    a[12] = (_Float16)(a3.x * sA); a[13] = (_Float16)(a3.y * sA);
    a[14] = (_Float16)(a3.z * sA); a[15] = (_Float16)(a3.w * sA);
    const v16h b = Bfrag[ks * 32];
    acc = __builtin_amdgcn_wmma_f32_16x16x32_f16(false, a, false, b,
                                                 (short)0, acc, false, false);
  }

  const int n = tileN + m16;     // D: lane holds column n, rows hi*8+j
#pragma unroll
  for (int j = 0; j < 8; ++j) {
    const int r = tileM + hi * 8 + j;
    if (n < ldc) {
      float v = 0.0f;
      if (n < N) { v = acc[j]; if (bias) v += bias[n]; }
      C[(size_t)r * ldc + n] = v;
    }
  }
}

// ------------------------------------------------------ graph scatter/fuse

// 8 edges per block; blockDim.x == feature width; E multiple of 8
__global__ void scatter_add(const float* __restrict__ y, int ldy,
                            const int* __restrict__ src, const int* __restrict__ dst,
                            float* __restrict__ agg, int ldagg) {
  const int c = threadIdx.x;
  const int e0 = blockIdx.x * 8;
#pragma unroll
  for (int i = 0; i < 8; ++i) {
    const int s = src[e0 + i], d = dst[e0 + i];
    atomicAdd(&agg[d * ldagg + c], y[s * ldy + c]);
  }
}

// out = relu(self + agg * rs_in[row] + bconv[col]), in place over `self`
__global__ void combine_relu(float* __restrict__ self_out, int lds,
                             const float* __restrict__ agg, int ldagg,
                             const float* __restrict__ rs_in,
                             const float* __restrict__ bconv) {
  const int i = blockIdx.x, j = threadIdx.x;
  float v = self_out[i * lds + j] + agg[i * ldagg + j] * rs_in[i] + bconv[j];
  self_out[i * lds + j] = v > 0.0f ? v : 0.0f;
}

// row L2-normalize [M,64] f32 -> f16; one wave per row, 2 elems per lane
__global__ void normalize_f16(const float* __restrict__ x, _Float16* __restrict__ out) {
  const int row = blockIdx.x;
  const int lane = threadIdx.x;          // 0..31
  float v0 = x[row * 64 + lane];
  float v1 = x[row * 64 + lane + 32];
  float s = v0 * v0 + v1 * v1;
#pragma unroll
  for (int o = 16; o > 0; o >>= 1) s += __shfl_xor(s, o, 32);
  const float inv = 1.0f / fmaxf(sqrtf(s), 1e-12f);
  out[row * 64 + lane]      = (_Float16)(v0 * inv);
  out[row * 64 + lane + 32] = (_Float16)(v1 * inv);
}

// ------------------------------------------- final rating GEMM (A @ B^T)
// C[16384,8192] f32 = An[16384,64]f16 @ (Bn[8192,64]f16)^T.  256 threads =
// 8 waves per block; block does a 128x128 C tile (wave = 16 rows x 128 cols
// = 8 WMMA tiles), K=64 in two 32-steps; fragments via global_load_b128.

__global__ __launch_bounds__(256) void rating_gemm(const _Float16* __restrict__ A,
                                                   const _Float16* __restrict__ B,
                                                   float* __restrict__ C, int ldc) {
  const int lane = threadIdx.x & 31;
  const int wave = threadIdx.x >> 5;      // 0..7
  const int m16 = lane & 15, hi = lane >> 4;
  const int rowBase = blockIdx.y * 128 + wave * 16;
  const int colBase = blockIdx.x * 128;
  const _Float16* Ap = A + (size_t)(rowBase + m16) * 64 + hi * 8;

  v8f acc[8] = {};
#pragma unroll
  for (int k0 = 0; k0 < 64; k0 += 32) {
    const v8h alo = *(const v8h*)(Ap + k0);
    const v8h ahi = *(const v8h*)(Ap + k0 + 16);
    const v16h a = __builtin_shufflevector(alo, ahi, 0, 1, 2, 3, 4, 5, 6, 7,
                                           8, 9, 10, 11, 12, 13, 14, 15);
#pragma unroll
    for (int nt = 0; nt < 8; ++nt) {
      const _Float16* Bq = B + (size_t)(colBase + nt * 16 + m16) * 64 + hi * 8;
      const v8h blo = *(const v8h*)(Bq + k0);
      const v8h bhi = *(const v8h*)(Bq + k0 + 16);
      const v16h b = __builtin_shufflevector(blo, bhi, 0, 1, 2, 3, 4, 5, 6, 7,
                                             8, 9, 10, 11, 12, 13, 14, 15);
      acc[nt] = __builtin_amdgcn_wmma_f32_16x16x32_f16(false, a, false, b,
                                                       (short)0, acc[nt], false, false);
    }
  }
#pragma unroll
  for (int nt = 0; nt < 8; ++nt) {
    const int n = colBase + nt * 16 + m16;
#pragma unroll
    for (int j = 0; j < 8; ++j) {
      const int r = rowBase + hi * 8 + j;
      C[(size_t)r * ldc + n] = acc[nt][j];
    }
  }
}

// ---------------------------------------------------------------- launcher

extern "C" void kernel_launch(void* const* d_in, const int* in_sizes, int n_in,
                              void* d_out, int out_size, void* d_ws, size_t ws_size,
                              hipStream_t stream) {
  (void)in_sizes; (void)n_in; (void)out_size; (void)ws_size;

  const int NU = 16384, NR = 8192, E = 262144;

  const float* user_feat = (const float*)d_in[0];
  const float* repo_feat = (const float*)d_in[1];
  const float* W_ue = (const float*)d_in[2];  const float* b_ue = (const float*)d_in[3];
  const float* W_re = (const float*)d_in[4];  const float* b_re = (const float*)d_in[5];
  const float* Wh_u2r = (const float*)d_in[6];  const float* bh_u2r = (const float*)d_in[7];
  const float* Wh_r2u = (const float*)d_in[8];  const float* bh_r2u = (const float*)d_in[9];
  const float* W_hcu = (const float*)d_in[10]; const float* b_hcu = (const float*)d_in[11];
  const float* W_hcr = (const float*)d_in[12]; const float* b_hcr = (const float*)d_in[13];
  const float* Wo_u2r = (const float*)d_in[14]; const float* bo_u2r = (const float*)d_in[15];
  const float* Wo_r2u = (const float*)d_in[16]; const float* bo_r2u = (const float*)d_in[17];
  const float* W_ocu = (const float*)d_in[18]; const float* b_ocu = (const float*)d_in[19];
  const float* W_ocr = (const float*)d_in[20]; const float* b_ocr = (const float*)d_in[21];
  const int* eu_s = (const int*)d_in[22];
  const int* eu_d = (const int*)d_in[23];
  const int* er_s = (const int*)d_in[24];
  const int* er_d = (const int*)d_in[25];

  float* ws = (float*)d_ws;
  // workspace arena (offsets in floats)
  const size_t o_rs    = 0;                               // 4 rsqrt arrays (49152)
  const size_t o_deg   = o_rs    + (size_t)(2*NU + 2*NR); // u32 counts     (49152)
  const size_t o_hU    = o_deg   + (size_t)(2*NU + 2*NR); // [NU,128]
  const size_t o_hR    = o_hU    + (size_t)NU * 128;      // [NR,128]
  const size_t o_y     = o_hR    + (size_t)NR * 128;      // [NU,96] msg / f16 later
  const size_t o_aggU  = o_y     + (size_t)NU * 96;       // [NU,96]
  const size_t o_aggR  = o_aggU  + (size_t)NU * 96;       // [NR,96] (contiguous after aggU)
  const size_t o_selfU = o_aggR  + (size_t)NR * 96;       // [NU,96] -> outU in place
  const size_t o_selfR = o_selfU + (size_t)NU * 96;       // [NR,96] -> outR in place
  const size_t o_bp    = o_selfR + (size_t)NR * 96;       // packed weight scratch (8192 f)

  float* rsUo = ws + o_rs;            // user out-degree (u2r src)
  float* rsUi = rsUo + NU;            // user in-degree  (r2u dst)
  float* rsRo = rsUi + NU;            // repo out-degree (r2u src)
  float* rsRi = rsRo + NR;            // repo in-degree  (u2r dst)
  unsigned* dg = (unsigned*)(ws + o_deg);
  unsigned* dgUo = dg; unsigned* dgUi = dgUo + NU;
  unsigned* dgRo = dgUi + NU; unsigned* dgRi = dgRo + NR;

  float* hU = ws + o_hU;   float* hR = ws + o_hR;
  float* y  = ws + o_y;
  float* aggU = ws + o_aggU; float* aggR = ws + o_aggR;
  float* selfU = ws + o_selfU; float* selfR = ws + o_selfR;
  _Float16* Bp = (_Float16*)(ws + o_bp);
  const long nDeg = 2L*NU + 2L*NR;
  const long nAgg = (long)NU*96 + (long)NR*96;   // contiguous aggU+aggR region

  // ---- degrees -> rsqrt(clip(deg,1))
  zero_f32<<<dim3((nDeg + 255) / 256), 256, 0, stream>>>((float*)dg, nDeg);
  degree_count<<<dim3((E + 255) / 256), 256, 0, stream>>>(eu_s, eu_d, E, dgUo, dgRi);
  degree_count<<<dim3((E + 255) / 256), 256, 0, stream>>>(er_s, er_d, E, dgRo, dgUi);
  deg_to_rsqrt<<<dim3((nDeg + 255) / 256), 256, 0, stream>>>(dg, ws + o_rs, (int)nDeg);

  // ---- embeddings: h = feat @ W + b   (K=64 -> ksteps 2, N=125 pad to 128)
  pack_b_f16<<<dim3(1), 8 * 2 * 32, 0, stream>>>(W_ue, 64, 125, Bp, 2);
  gemm_wmma<<<dim3(8, NU / 16), 32, 0, stream>>>(user_feat, 64, nullptr, Bp, 2, b_ue,
                                                 hU, 128, NU, 125);
  pack_b_f16<<<dim3(1), 8 * 2 * 32, 0, stream>>>(W_re, 64, 125, Bp, 2);
  gemm_wmma<<<dim3(8, NR / 16), 32, 0, stream>>>(repo_feat, 64, nullptr, Bp, 2, b_re,
                                                 hR, 128, NR, 125);

  // ---- level 1 graph conv (K=125 packed, A padded to 128 -> ksteps 4)
  zero_f32<<<dim3((nAgg + 255) / 256), 256, 0, stream>>>(aggU, nAgg);
  pack_b_f16<<<dim3(1), 6 * 4 * 32, 0, stream>>>(Wh_u2r, 125, 96, Bp, 4);
  gemm_wmma<<<dim3(6, NU / 16), 32, 0, stream>>>(hU, 128, rsUo, Bp, 4, nullptr,
                                                 y, 96, NU, 96);
  scatter_add<<<dim3(E / 8), 96, 0, stream>>>(y, 96, eu_s, eu_d, aggR, 96);
  pack_b_f16<<<dim3(1), 6 * 4 * 32, 0, stream>>>(Wh_r2u, 125, 96, Bp, 4);
  gemm_wmma<<<dim3(6, NR / 16), 32, 0, stream>>>(hR, 128, rsRo, Bp, 4, nullptr,
                                                 y, 96, NR, 96);
  scatter_add<<<dim3(E / 8), 96, 0, stream>>>(y, 96, er_s, er_d, aggU, 96);
  pack_b_f16<<<dim3(1), 6 * 4 * 32, 0, stream>>>(W_hcu, 125, 96, Bp, 4);
  gemm_wmma<<<dim3(6, NU / 16), 32, 0, stream>>>(hU, 128, nullptr, Bp, 4, b_hcu,
                                                 selfU, 96, NU, 96);
  pack_b_f16<<<dim3(1), 6 * 4 * 32, 0, stream>>>(W_hcr, 125, 96, Bp, 4);
  gemm_wmma<<<dim3(6, NR / 16), 32, 0, stream>>>(hR, 128, nullptr, Bp, 4, b_hcr,
                                                 selfR, 96, NR, 96);
  combine_relu<<<dim3(NU), 96, 0, stream>>>(selfU, 96, aggU, 96, rsUi, bh_r2u); // -> outU
  combine_relu<<<dim3(NR), 96, 0, stream>>>(selfR, 96, aggR, 96, rsRi, bh_u2r); // -> outR

  // ---- level 2 graph conv (K=96 -> ksteps 3); reuse agg/y, hU/hR as self2
  zero_f32<<<dim3((nAgg + 255) / 256), 256, 0, stream>>>(aggU, nAgg);
  pack_b_f16<<<dim3(1), 4 * 3 * 32, 0, stream>>>(Wo_u2r, 96, 64, Bp, 3);
  gemm_wmma<<<dim3(4, NU / 16), 32, 0, stream>>>(selfU, 96, rsUo, Bp, 3, nullptr,
                                                 y, 64, NU, 64);
  scatter_add<<<dim3(E / 8), 64, 0, stream>>>(y, 64, eu_s, eu_d, aggR, 64);
  pack_b_f16<<<dim3(1), 4 * 3 * 32, 0, stream>>>(Wo_r2u, 96, 64, Bp, 3);
  gemm_wmma<<<dim3(4, NR / 16), 32, 0, stream>>>(selfR, 96, rsRo, Bp, 3, nullptr,
                                                 y, 64, NR, 64);
  scatter_add<<<dim3(E / 8), 64, 0, stream>>>(y, 64, er_s, er_d, aggU, 64);
  pack_b_f16<<<dim3(1), 4 * 3 * 32, 0, stream>>>(W_ocu, 96, 64, Bp, 3);
  gemm_wmma<<<dim3(4, NU / 16), 32, 0, stream>>>(selfU, 96, nullptr, Bp, 3, b_ocu,
                                                 hU, 64, NU, 64);
  pack_b_f16<<<dim3(1), 4 * 3 * 32, 0, stream>>>(W_ocr, 96, 64, Bp, 3);
  gemm_wmma<<<dim3(4, NR / 16), 32, 0, stream>>>(selfR, 96, nullptr, Bp, 3, b_ocr,
                                                 hR, 64, NR, 64);
  combine_relu<<<dim3(NU), 64, 0, stream>>>(hU, 64, aggU, 64, rsUi, bo_r2u); // -> h_user_new
  combine_relu<<<dim3(NR), 64, 0, stream>>>(hR, 64, aggR, 64, rsRi, bo_u2r); // -> h_repo_new

  // ---- normalize to f16 (y region is free now)
  _Float16* AnU = (_Float16*)y;
  _Float16* AnR = AnU + (size_t)NU * 64;
  normalize_f16<<<dim3(NU), 32, 0, stream>>>(hU, AnU);
  normalize_f16<<<dim3(NR), 32, 0, stream>>>(hR, AnR);

  // ---- ratings = AnU @ AnR^T  -> [16384, 8192] f32
  rating_gemm<<<dim3(NR / 128, NU / 128), 256, 0, stream>>>(AnU, AnR, (float*)d_out, NR);
}
